// myLSTM_69045894250891
// MI455X (gfx1250) — matrix operand order repaired and checked by reading, not compiled, not run on hardware
//
#include <hip/hip_runtime.h>

// ---------------------------------------------------------------------------
// CDNA5 (gfx1250) LSTM: bf16 WMMA GEMMs + persistent recurrent scan.
// B=32, T=2048, D=128, H=256, gates G4 -> N = 4H = 1024.
// ---------------------------------------------------------------------------

typedef __attribute__((ext_vector_type(16))) __bf16 v16bf;
typedef __attribute__((ext_vector_type(8)))  float  v8f;
typedef __attribute__((ext_vector_type(4)))  float  f32x4;  // native vec for NT builtins

union Frag {
  v16bf v;
  uint4 q[2];
  unsigned int u[8];
};

// Native converts: let clang lower (__bf16)f casts to v_cvt_*bf16_f32.
__device__ __forceinline__ unsigned int packbf2(float lo, float hi) {
  union { __bf16 b[2]; unsigned int u; } t;
  t.b[0] = (__bf16)lo;
  t.b[1] = (__bf16)hi;
  return t.u;
}

__device__ __forceinline__ v16bf cvt_frag(float4 c0, float4 c1, float4 c2, float4 c3) {
  v16bf r;
  r[0]  = (__bf16)c0.x; r[1]  = (__bf16)c0.y; r[2]  = (__bf16)c0.z; r[3]  = (__bf16)c0.w;
  r[4]  = (__bf16)c1.x; r[5]  = (__bf16)c1.y; r[6]  = (__bf16)c1.z; r[7]  = (__bf16)c1.w;
  r[8]  = (__bf16)c2.x; r[9]  = (__bf16)c2.y; r[10] = (__bf16)c2.z; r[11] = (__bf16)c2.w;
  r[12] = (__bf16)c3.x; r[13] = (__bf16)c3.y; r[14] = (__bf16)c3.z; r[15] = (__bf16)c3.w;
  return r;
}

__device__ __forceinline__ float sigmoidf_(float x) {
  return 1.0f / (1.0f + __expf(-x));
}
__device__ __forceinline__ float tanh_(float x) {
  float e = __expf(-2.0f * x);
  return (1.0f - e) / (1.0f + e);
}

// ---------------------------------------------------------------------------
// Packed weight layouts (lane-fragment order so loads are 2x b128 per lane):
//   packW: [ntile(64)][ktile(4)][lane(32)][8 dwords]   (W = [128 x 1024] bf16)
//   packU: [ntile(64)][ktile(8)][lane(32)][8 dwords]   (U = [256 x 1024] bf16)
// B-matrix fragment (16-bit, KxN = 32x16): lane n (0..15) holds col n,
// K = kt*32 + 0..7, 16..23 (2 per dword); lane n+16 holds col n, K = kt*32 + 8..15, 24..31.
// ---------------------------------------------------------------------------
#define NPW (64 * 4 * 32 * 8)
#define NPU (64 * 8 * 32 * 8)

__global__ void lstm_pack_weights(
    const float* __restrict__ wf, const float* __restrict__ wi,
    const float* __restrict__ wo, const float* __restrict__ wc,
    const float* __restrict__ uf, const float* __restrict__ ui,
    const float* __restrict__ uo, const float* __restrict__ uc,
    unsigned int* __restrict__ packW, unsigned int* __restrict__ packU) {
  int idx = blockIdx.x * blockDim.x + threadIdx.x;
  if (idx < NPW) {
    int d    = idx & 7;
    int lane = (idx >> 3) & 31;
    int kt   = (idx >> 8) & 3;
    int nt   = idx >> 10;
    int col  = nt * 16 + (lane & 15);
    int k    = kt * 32 + (d >> 2) * 16 + (lane >> 4) * 8 + (d & 3) * 2;
    const float* w = (col < 256) ? wf : (col < 512) ? wi : (col < 768) ? wo : wc;
    int c0 = col & 255;
    packW[idx] = packbf2(w[(size_t)k * 256 + c0], w[(size_t)(k + 1) * 256 + c0]);
  } else if (idx < NPW + NPU) {
    int iu   = idx - NPW;
    int d    = iu & 7;
    int lane = (iu >> 3) & 31;
    int kt   = (iu >> 8) & 7;
    int nt   = iu >> 11;
    int col  = nt * 16 + (lane & 15);
    int k    = kt * 32 + (d >> 2) * 16 + (lane >> 4) * 8 + (d & 3) * 2;
    const float* u = (col < 256) ? uf : (col < 512) ? ui : (col < 768) ? uo : uc;
    int c0 = col & 255;
    packU[iu] = packbf2(u[(size_t)k * 256 + c0], u[(size_t)(k + 1) * 256 + c0]);
  }
}

// ---------------------------------------------------------------------------
// Phase 1: xg[M=65536, N=1024] = x[M,128] @ W + bias (bias folded here).
// Grid (4096, 4), block 128 (4 waves). Each wave: one M-tile x 4 N-tiles, K=128.
// A fragment (16-bit, MxK = 16x32): lane m (0..15) holds row m,
//   chunk0 K = kt*32 + 8*(lane>>4) .. +7, chunk1 = chunk0 + 16.
// ---------------------------------------------------------------------------
__global__ void lstm_xgemm(
    const float* __restrict__ x, const unsigned int* __restrict__ packW,
    const float* __restrict__ bf, const float* __restrict__ bi,
    const float* __restrict__ bo, const float* __restrict__ bc,
    float* __restrict__ xg) {
  const int mtile = blockIdx.x;                   // 0..4095
  const int lane  = threadIdx.x & 31;
  const int wave  = threadIdx.x >> 5;             // 0..3
  const int ntb   = (blockIdx.y * 4 + wave) * 4;  // base of 4 n-tiles
  const int g     = lane >> 4;
  const int row   = mtile * 16 + (lane & 15);

  // Load A fragments (x rows, f32 -> bf16), 4 k-tiles.
  Frag A[4];
#pragma unroll
  for (int kt = 0; kt < 4; ++kt) {
    const float4* p0 = (const float4*)(x + (size_t)row * 128 + kt * 32 + g * 8);
    const float4* p1 = (const float4*)(x + (size_t)row * 128 + kt * 32 + g * 8 + 16);
    A[kt].v = cvt_frag(p0[0], p0[1], p1[0], p1[1]);
  }

  v8f acc[4];
#pragma unroll
  for (int n = 0; n < 4; ++n) {
    acc[n] = (v8f){};
    const int nt = ntb + n;
#pragma unroll
    for (int kt = 0; kt < 4; ++kt) {
      Frag B;
      const uint4* pb = (const uint4*)(packW + (size_t)((nt * 4 + kt) * 32 + lane) * 8);
      B.q[0] = pb[0];
      B.q[1] = pb[1];
      acc[n] = __builtin_amdgcn_wmma_f32_16x16x32_bf16(
          false, A[kt].v, false, B.v, (short)0, acc[n], false, false);
    }
  }

#pragma unroll
  for (int n = 0; n < 4; ++n) {
    const int nt  = ntb + n;
    const int col = nt * 16 + (lane & 15);
    const float* bptr = (col < 256) ? bf : (col < 512) ? bi : (col < 768) ? bo : bc;
    const float bias = bptr[col & 255];
#pragma unroll
    for (int r = 0; r < 8; ++r) {
      int m = mtile * 16 + r + 8 * g;
      // xg is a 256 MiB single-use stream: keep it out of the way of L2.
      __builtin_nontemporal_store(acc[n][r] + bias, &xg[(size_t)m * 1024 + col]);
    }
  }
}

// ---------------------------------------------------------------------------
// Phase 2: persistent recurrent scan. 2 blocks x 512 threads (16 waves).
// Block bg owns batch rows [16*bg, 16*bg+16). U is register-resident bf16.
// Per step: A(h) from LDS -> 512 WMMA -> g to LDS -> gates/state -> h to LDS.
// ---------------------------------------------------------------------------
#define HS 264   // h-buffer row stride (bf16 elems): 528B = 132 dwords (conflict-friendly)
#define GS 1028  // g-buffer row stride (f32 elems): +4 dwords pad

__global__ void __launch_bounds__(512, 1) lstm_scan(
    const float* __restrict__ xg, const unsigned int* __restrict__ packU,
    float* __restrict__ out) {
  __shared__ unsigned short hs_[16 * HS];  // h (bf16), padded rows
  __shared__ float gbuf[16 * GS];          // h@U partial gates (f32)

  const int tid  = threadIdx.x;
  const int lane = tid & 31;
  const int wave = tid >> 5;               // 0..15
  const int bg   = blockIdx.x;             // 0..1
  const int g    = lane >> 4;
  const int arow = lane & 15;

  // Register-resident U fragments: 4 n-tiles x 8 k-tiles per wave.
  Frag B[4][8];
#pragma unroll
  for (int n = 0; n < 4; ++n) {
#pragma unroll
    for (int kt = 0; kt < 8; ++kt) {
      const uint4* pb =
          (const uint4*)(packU + (size_t)((((wave * 4 + n) * 8 + kt) * 32) + lane) * 8);
      B[n][kt].q[0] = pb[0];
      B[n][kt].q[1] = pb[1];
    }
  }

  // h0 = 0
  for (int i = tid; i < 16 * HS; i += 512) hs_[i] = 0;

  // Elementwise ownership: row = tid/32, 8 consecutive cols.
  const int erow  = tid >> 5;
  const int ecolb = (tid & 31) * 8;
  float c[8];
#pragma unroll
  for (int i = 0; i < 8; ++i) c[i] = 0.0f;

  const float* xg_base = xg + (size_t)(bg * 16 + erow) * 2048 * 1024;
  float* out_base = out + ((size_t)(bg * 16 + erow) * 2048) * 256 + ecolb;

  __syncthreads();

  for (int t = 0; t < 2048; ++t) {
    // ---- A fragments (h, bf16) from LDS ----
    Frag A[8];
#pragma unroll
    for (int kt = 0; kt < 8; ++kt) {
      int kb = kt * 32 + g * 8;
      A[kt].q[0] = *(const uint4*)(hs_ + arow * HS + kb);
      A[kt].q[1] = *(const uint4*)(hs_ + arow * HS + kb + 16);
    }

    // ---- h @ U : 4 n-tiles x 8 k-tiles of v_wmma_f32_16x16x32_bf16 ----
#pragma unroll
    for (int n = 0; n < 4; ++n) {
      v8f acc = (v8f){};
#pragma unroll
      for (int kt = 0; kt < 8; ++kt) {
        acc = __builtin_amdgcn_wmma_f32_16x16x32_bf16(
            false, A[kt].v, false, B[n][kt].v, (short)0, acc, false, false);
      }
      const int col = (wave * 4 + n) * 16 + (lane & 15);
#pragma unroll
      for (int r = 0; r < 8; ++r) gbuf[(r + 8 * g) * GS + col] = acc[r];
    }
    __syncthreads();

    // ---- gates + state update (f32), xg streamed from global ----
    const float* xgrow = xg_base + (size_t)t * 1024;
    // Prefetch next timestep's activations.
    if (t + 1 < 2048) {
#pragma unroll
      for (int G = 0; G < 4; ++G)
        __builtin_prefetch(xgrow + 1024 + G * 256 + ecolb, 0, 1);
    }

    float gv[4][8];
#pragma unroll
    for (int G = 0; G < 4; ++G) {
      f32x4 a0 = *(const f32x4*)(gbuf + erow * GS + G * 256 + ecolb);
      f32x4 a1 = *(const f32x4*)(gbuf + erow * GS + G * 256 + ecolb + 4);
      f32x4 x0 = __builtin_nontemporal_load((const f32x4*)(xgrow + G * 256 + ecolb));
      f32x4 x1 = __builtin_nontemporal_load((const f32x4*)(xgrow + G * 256 + ecolb + 4));
      gv[G][0] = a0.x + x0.x; gv[G][1] = a0.y + x0.y;
      gv[G][2] = a0.z + x0.z; gv[G][3] = a0.w + x0.w;
      gv[G][4] = a1.x + x1.x; gv[G][5] = a1.y + x1.y;
      gv[G][6] = a1.z + x1.z; gv[G][7] = a1.w + x1.w;
    }

    float h[8];
#pragma unroll
    for (int i = 0; i < 8; ++i) {
      float fg = sigmoidf_(gv[0][i]);
      float ig = sigmoidf_(gv[1][i]);
      float og = sigmoidf_(gv[2][i]);
      float cd = tanh_(gv[3][i]);
      float cn = fg * c[i] + ig * cd;
      c[i] = cn;
      h[i] = og * tanh_(cn);
    }

    // h (f32) -> output (write-only stream: nontemporal)
    float* op = out_base + (size_t)t * 256;
    __builtin_nontemporal_store((f32x4){h[0], h[1], h[2], h[3]}, (f32x4*)op);
    __builtin_nontemporal_store((f32x4){h[4], h[5], h[6], h[7]}, (f32x4*)(op + 4));

    // h (bf16) -> LDS for next step's A
    union { __bf16 b[8]; uint4 q; } hb;
#pragma unroll
    for (int i = 0; i < 8; ++i) hb.b[i] = (__bf16)h[i];
    *(uint4*)(hs_ + erow * HS + ecolb) = hb.q;

    __syncthreads();
  }
}

// ---------------------------------------------------------------------------
extern "C" void kernel_launch(void* const* d_in, const int* in_sizes, int n_in,
                              void* d_out, int out_size, void* d_ws, size_t ws_size,
                              hipStream_t stream) {
  const float* x  = (const float*)d_in[0];
  const float* wf = (const float*)d_in[1];
  const float* wi = (const float*)d_in[2];
  const float* wo = (const float*)d_in[3];
  const float* wc = (const float*)d_in[4];
  const float* uf = (const float*)d_in[5];
  const float* ui = (const float*)d_in[6];
  const float* uo = (const float*)d_in[7];
  const float* uc = (const float*)d_in[8];
  const float* bf = (const float*)d_in[9];
  const float* bi = (const float*)d_in[10];
  const float* bo = (const float*)d_in[11];
  const float* bc = (const float*)d_in[12];
  float* out = (float*)d_out;

  char* ws = (char*)d_ws;
  const size_t XG_BYTES = (size_t)32 * 2048 * 1024 * sizeof(float);  // 256 MiB
  float* xg = (float*)ws;
  unsigned int* packW = (unsigned int*)(ws + XG_BYTES);
  unsigned int* packU = packW + NPW;

  // 1) Pack W/U into WMMA bf16 lane-fragment layout.
  {
    int total = NPW + NPU;
    lstm_pack_weights<<<(total + 255) / 256, 256, 0, stream>>>(
        wf, wi, wo, wc, uf, ui, uo, uc, packW, packU);
  }
  // 2) xg = x @ W + b  (WMMA bf16, f32 accumulate).
  {
    dim3 grid(4096, 4);
    lstm_xgemm<<<grid, 128, 0, stream>>>(x, packW, bf, bi, bo, bc, xg);
  }
  // 3) Recurrent scan: 2 persistent workgroups (16 batch rows each).
  lstm_scan<<<2, 512, 0, stream>>>(xg, packU, out);
}
